// UberCRSN_24902220382520
// MI455X (gfx1250) — matrix operand order repaired and checked by reading, not compiled, not run on hardware
//
#include <hip/hip_runtime.h>
#include <hip/hip_bf16.h>
#include <math.h>

// ---------------------------------------------------------------------------
// UberCRSN forward for MI455X (gfx1250, wave32, WMMA).
// All GEMMs use V_WMMA_F32_16X16X32_F16 (f16 inputs, f32 accumulate).
// All WMMA operand loads are contiguous (weights pre-transposed, K stored
// transposed, f16 activation mirrors) -> b128-class A loads; B loads are strided u16 gathers along K.
// Attention is fused flash-style (never materializes the 4096x4096 matrix).
// Requires ws_size >= ~92 MB (dominated by the 4096x16x256 stack memory).
// ---------------------------------------------------------------------------

typedef __attribute__((ext_vector_type(16))) _Float16 v16h;
typedef __attribute__((ext_vector_type(8)))  float    v8f;

#define BSZ   4096
#define DDIM  128
#define SLOTS 16
#define NSYMS 128
#define TSTEP 8
#define EPSF  1e-6f
#define ACTTH 0.9999f
#define GBSF  0.8f
#define CCF   0.01f

// ------------------------- WMMA helpers (layout per ISA 7.12.2) -------------
__device__ __forceinline__ v8f wmma16(v16h a, v16h b, v8f c) {
  return __builtin_amdgcn_wmma_f32_16x16x32_f16(false, a, false, b, (short)0, c,
                                                false, false);
}
__device__ __forceinline__ v8f vzero8() {
  v8f c;
#pragma unroll
  for (int i = 0; i < 8; ++i) c[i] = 0.f;
  return c;
}
// A fragment: rows m0..m0+15, K = k0..k0+31. lane<16 holds K k0+0..7 / k0+16..23,
// lane>=16 holds K k0+8..15 / k0+24..31 (same M rows). Contiguous 8-half runs.
__device__ __forceinline__ v16h load_a_f16(const _Float16* X, int ld, int m0, int k0) {
  int lane = threadIdx.x & 31;
  const _Float16* p = X + (size_t)(m0 + (lane & 15)) * ld + k0 + ((lane >> 4) << 3);
  v16h a;
#pragma unroll
  for (int i = 0; i < 8; ++i) a[i] = p[i];
#pragma unroll
  for (int i = 0; i < 8; ++i) a[8 + i] = p[16 + i];
  return a;
}
__device__ __forceinline__ v16h load_a_f32(const float* X, int ld, int m0, int k0) {
  int lane = threadIdx.x & 31;
  const float* p = X + (size_t)(m0 + (lane & 15)) * ld + k0 + ((lane >> 4) << 3);
  v16h a;
#pragma unroll
  for (int i = 0; i < 8; ++i) a[i] = (_Float16)p[i];
#pragma unroll
  for (int i = 0; i < 8; ++i) a[8 + i] = (_Float16)p[16 + i];
  return a;
}
// B fragment, K x N matrix (row-major, contiguous N): B[k][n] = X[k0+k][n0+n].
// Lane holds column n0+(lane&15); K in table order (strided gather along K).
__device__ __forceinline__ v16h load_b_f16(const _Float16* X, int ld, int k0, int n0) {
  int lane = threadIdx.x & 31;
  const _Float16* p = X + (size_t)(k0 + ((lane >> 4) << 3)) * ld + n0 + (lane & 15);
  v16h b;
#pragma unroll
  for (int i = 0; i < 8; ++i) b[i] = p[(size_t)i * ld];
#pragma unroll
  for (int i = 0; i < 8; ++i) b[8 + i] = p[(size_t)(16 + i) * ld];
  return b;
}

__device__ __forceinline__ float wsum32(float v) {
#pragma unroll
  for (int o = 16; o > 0; o >>= 1) v += __shfl_xor(v, o, 32);
  return v;
}
__device__ __forceinline__ float wmax32(float v) {
#pragma unroll
  for (int o = 16; o > 0; o >>= 1) v = fmaxf(v, __shfl_xor(v, o, 32));
  return v;
}
__device__ __forceinline__ float sigm(float x) { return 1.f / (1.f + expf(-x)); }

// ------------------------- prep: f32 weights -> TRANSPOSED f16 copies -------
// All weights stored as W^T in [K][N] layout so B-fragments are contiguous.
// wbuf half-offsets: cellWr 0, cellWi 16384, Wq_r 32768, Wq_i 49152,
// Wk_r 65536, Wk_i 81920, Wv_r 98304, Wv_i 114688, codebook^T 131072 ([256][128]),
// Wd^T 163840 ([256][128]).
__global__ void prep_kernel(const float* cwr, const float* cwi,
                            const float* wqr, const float* wqi,
                            const float* wkr, const float* wki,
                            const float* wvr, const float* wvi,
                            const float* cb,  const float* wd,
                            _Float16* wbuf, float* code_sq) {
  int tid = blockIdx.x * blockDim.x + threadIdx.x;
  int n = gridDim.x * blockDim.x;
  const float* srcs[8] = {cwr, cwi, wqr, wqi, wkr, wki, wvr, wvi};
  for (int i = tid; i < 8 * 16384; i += n) {
    int s = i >> 14, e = i & 16383;
    int k = e >> 7, c = e & 127;                 // wT[k][c] = W[c][k]
    wbuf[i] = (_Float16)srcs[s][c * 128 + k];
  }
  for (int i = tid; i < 32768; i += n) {
    int k = i >> 7, j = i & 127;                 // [256][128] transposed
    wbuf[131072 + i] = (_Float16)cb[j * 256 + k];
    wbuf[163840 + i] = (_Float16)wd[j * 256 + k];
  }
  for (int j = tid; j < NSYMS; j += n) {
    float s = 0.f;
    for (int k = 0; k < 2 * DDIM; ++k) { float v = cb[j * 256 + k]; s += v * v; }
    code_sq[j] = s;
  }
}

// ------------------------- init: embedding + state --------------------------
__global__ void init_kernel(const int* ids, const float* emag, const float* ephase,
                            _Float16* zrh, _Float16* zih, float* zw, float* stck,
                            float* ptrb, float* halting, float* remain, float* vqt,
                            float* etht, float* ponder, int* cursym) {
  int tid = blockIdx.x * blockDim.x + threadIdx.x;
  int n = gridDim.x * blockDim.x;
  for (int i = tid; i < BSZ * DDIM; i += n) {
    int b = i >> 7, d = i & 127;
    int v = ids[b];
    float r = emag[v * DDIM + d], th = ephase[v * DDIM + d];
    zrh[i] = (_Float16)(r * cosf(th));
    zih[i] = (_Float16)(r * sinf(th));
  }
  for (int i = tid; i < BSZ * 256; i += n) zw[i] = 0.f;
  for (size_t i = tid; i < (size_t)BSZ * SLOTS * 256; i += n) stck[i] = 0.f;
  for (int i = tid; i < BSZ * SLOTS; i += n) ptrb[i] = ((i & 15) == 0) ? 1.f : 0.f;
  for (int i = tid; i < BSZ; i += n) {
    halting[i] = 0.f; remain[i] = 1.f; vqt[i] = 0.f;
    etht[i] = 0.f; ponder[i] = 0.f; cursym[i] = -1;
  }
}

// ------------------------- cell: clin + cnorm + modrelu ---------------------
// block = 256 threads (8 waves). Block owns 16 rows; wave w owns N-tile w*16.
__global__ void cell_kernel(const _Float16* zrh, const _Float16* zih,
                            const _Float16* wrT, const _Float16* wiT,
                            const float* ln_scale, const float* ln_shift,
                            const float* mod_bias, _Float16* prh, _Float16* pih) {
  __shared__ float sR[16][128], sI[16][128], sMean[16], sRstd[16];
  int wave = threadIdx.x >> 5, lane = threadIdx.x & 31;
  int m0 = blockIdx.x * 16, n0 = wave * 16;
  v8f rr = vzero8(), ii = vzero8(), ir = vzero8(), ri = vzero8();
  for (int kt = 0; kt < 4; ++kt) {
    int k = kt * 32;
    v16h axr = load_a_f16(zrh, DDIM, m0, k);
    v16h axi = load_a_f16(zih, DDIM, m0, k);
    v16h bwr = load_b_f16(wrT, DDIM, k, n0);
    v16h bwi = load_b_f16(wiT, DDIM, k, n0);
    rr = wmma16(axr, bwr, rr);
    ii = wmma16(axi, bwi, ii);
    ir = wmma16(axi, bwr, ir);
    ri = wmma16(axr, bwi, ri);
  }
  int r0 = (lane >> 4) << 3, cc = n0 + (lane & 15);
#pragma unroll
  for (int j = 0; j < 8; ++j) {
    sR[r0 + j][cc] = rr[j] - ii[j];
    sI[r0 + j][cc] = ir[j] + ri[j];
  }
  __syncthreads();
  if (threadIdx.x < 16) {
    int r = threadIdx.x;
    float s = 0.f, s2 = 0.f;
    for (int c = 0; c < 128; ++c) {
      float a = sR[r][c], b = sI[r][c];
      float mag = sqrtf(a * a + b * b) + EPSF;
      s += mag; s2 += mag * mag;
    }
    float mean = s / 128.f;
    float var = (s2 - 128.f * mean * mean) / 127.f;   // ddof=1
    sMean[r] = mean;
    sRstd[r] = rsqrtf(var + EPSF);
  }
  __syncthreads();
  for (int e = threadIdx.x; e < 2048; e += 256) {
    int r = e >> 7, c = e & 127;
    float a = sR[r][c], b = sI[r][c];
    float m0v = sqrtf(a * a + b * b);
    float mag = m0v + EPSF;
    float nm = (mag - sMean[r]) * sRstd[r] * ln_scale[c] + ln_shift[c];
    float cp = (m0v > 0.f) ? a / m0v : 1.f;  // cos(atan2) / sin(atan2)
    float sp = (m0v > 0.f) ? b / m0v : 0.f;
    float nr = nm * cp, ni = nm * sp;
    float nn = sqrtf(nr * nr + ni * ni) + EPSF;
    float sc = fmaxf(nn + mod_bias[c], 0.f) / nn;     // modrelu
    prh[(size_t)(m0 + r) * DDIM + c] = (_Float16)(nr * sc);
    pih[(size_t)(m0 + r) * DDIM + c] = (_Float16)(ni * sc);
  }
}

// ------------------------- qkv: three complex linears -----------------------
// grid = (256 M-tiles, 3). Q/V written [B][256] (real 0..127, imag 128..255);
// K written transposed khT [256][4096] so attention B-fragments are contiguous.
__global__ void qkv_kernel(const _Float16* prh, const _Float16* pih,
                           const _Float16* wbuf, _Float16* qh, _Float16* khT,
                           _Float16* vh) {
  int which = blockIdx.y;
  const _Float16* wrT = wbuf + 32768 + which * 32768;
  const _Float16* wiT = wrT + 16384;
  int wave = threadIdx.x >> 5, lane = threadIdx.x & 31;
  int m0 = blockIdx.x * 16, n0 = wave * 16;
  v8f rr = vzero8(), ii = vzero8(), ir = vzero8(), ri = vzero8();
  for (int kt = 0; kt < 4; ++kt) {
    int k = kt * 32;
    v16h axr = load_a_f16(prh, DDIM, m0, k);
    v16h axi = load_a_f16(pih, DDIM, m0, k);
    v16h bwr = load_b_f16(wrT, DDIM, k, n0);
    v16h bwi = load_b_f16(wiT, DDIM, k, n0);
    rr = wmma16(axr, bwr, rr);
    ii = wmma16(axi, bwi, ii);
    ir = wmma16(axi, bwr, ir);
    ri = wmma16(axr, bwi, ri);
  }
  int r0 = (lane >> 4) << 3, nn = n0 + (lane & 15);
  if (which == 1) {  // K transposed: contiguous along j (rows)
    size_t rowb = (size_t)(m0 + r0);
#pragma unroll
    for (int j = 0; j < 8; ++j) {
      khT[(size_t)nn * BSZ + rowb + j]         = (_Float16)(rr[j] - ii[j]);
      khT[(size_t)(128 + nn) * BSZ + rowb + j] = (_Float16)(ir[j] + ri[j]);
    }
  } else {
    _Float16* dst = (which == 0) ? qh : vh;
#pragma unroll
    for (int j = 0; j < 8; ++j) {
      size_t row = (size_t)(m0 + r0 + j) * 256;
      dst[row + nn]       = (_Float16)(rr[j] - ii[j]);
      dst[row + 128 + nn] = (_Float16)(ir[j] + ri[j]);
    }
  }
}

// ------------------------- flash attention over the batch -------------------
// block = 8 waves per 16-row query tile; wave w owns KV stripe {it*256+w*32}.
__global__ void attn_kernel(const _Float16* qh, const _Float16* khT,
                            const _Float16* vh, float* af) {
  __shared__ float    sS[8][16][32];
  __shared__ _Float16 sP[8][16][32];
  __shared__ float    sM[8][16], sL[8][16], sC[8][16];
  __shared__ float    sOut[16][256];
  int wave = threadIdx.x >> 5, lane = threadIdx.x & 31;
  int m0 = blockIdx.x * 16;
  const float scale = 0.08838834764831845f;  // D^-0.5
  v8f acc[16];
  for (int jt = 0; jt < 16; ++jt) acc[jt] = vzero8();
  v16h aq[8];  // Q fragments are invariant over the KV loop: hoist
#pragma unroll
  for (int kt = 0; kt < 8; ++kt) aq[kt] = load_a_f16(qh, 256, m0, kt * 32);
  if (lane < 16) { sM[wave][lane] = -3.0e38f; sL[wave][lane] = 0.f; }
  __syncthreads();
  int r0 = (lane >> 4) << 3, nn = lane & 15;
  for (int it = 0; it < 16; ++it) {
    int kv0 = it * 256 + wave * 32;
    v8f c0 = vzero8(), c1 = vzero8();
    for (int kt = 0; kt < 8; ++kt) {
      int k = kt * 32;
      v16h b0 = load_b_f16(khT, BSZ, k, kv0);       // contiguous 32B reads
      v16h b1 = load_b_f16(khT, BSZ, k, kv0 + 16);
      c0 = wmma16(aq[kt], b0, c0);
      c1 = wmma16(aq[kt], b1, c1);
    }
#pragma unroll
    for (int j = 0; j < 8; ++j) {
      sS[wave][r0 + j][nn]      = c0[j] * scale;
      sS[wave][r0 + j][16 + nn] = c1[j] * scale;
    }
    __syncthreads();
    if (lane < 16) {  // online softmax, one row per lane
      int r = lane;
      float mo = sM[wave][r], tm = mo;
      for (int i = 0; i < 32; ++i) tm = fmaxf(tm, sS[wave][r][i]);
      float corr = expf(mo - tm);
      float sum = 0.f;
      for (int i = 0; i < 32; ++i) {
        float pv = expf(sS[wave][r][i] - tm);
        sum += pv;
        sP[wave][r][i] = (_Float16)pv;
      }
      sL[wave][r] = sL[wave][r] * corr + sum;
      sM[wave][r] = tm;
      sC[wave][r] = corr;
    }
    __syncthreads();
    float cj[8];
#pragma unroll
    for (int j = 0; j < 8; ++j) cj[j] = sC[wave][r0 + j];
    for (int jt = 0; jt < 16; ++jt)
#pragma unroll
      for (int j = 0; j < 8; ++j) acc[jt][j] *= cj[j];
    v16h pa = load_a_f16(&sP[wave][0][0], 32, 0, 0);
    for (int jt = 0; jt < 16; ++jt) {
      v16h bv = load_b_f16(vh, 256, kv0, jt * 16);
      acc[jt] = wmma16(pa, bv, acc[jt]);
    }
  }
  __syncthreads();
  if (threadIdx.x < 16) {  // merge the 8 per-wave softmax states
    int r = threadIdx.x;
    float gm = -3.0e38f;
    for (int w = 0; w < 8; ++w) gm = fmaxf(gm, sM[w][r]);
    float gl = 0.f;
    for (int w = 0; w < 8; ++w) gl += sL[w][r] * expf(sM[w][r] - gm);
    for (int w = 0; w < 8; ++w) sC[w][r] = expf(sM[w][r] - gm) / gl;
  }
  __syncthreads();
  for (int e = threadIdx.x; e < 4096; e += 256) ((float*)sOut)[e] = 0.f;
  __syncthreads();
  for (int wv = 0; wv < 8; ++wv) {
    if (wave == wv) {
      for (int jt = 0; jt < 16; ++jt)
#pragma unroll
        for (int j = 0; j < 8; ++j)
          sOut[r0 + j][jt * 16 + nn] += acc[jt][j] * sC[wave][r0 + j];
    }
    __syncthreads();
  }
  for (int e = threadIdx.x; e < 4096; e += 256) {
    int r = e >> 8, c = e & 255;
    af[(size_t)(m0 + r) * 256 + c] = sOut[r][c];
  }
}

// ------------------------- stack / controller (one wave per row) ------------
__global__ void stack_kernel(const float* af, const float* Wh, const float* bh,
                             const float* Wsm, const float* bs, float* stck,
                             float* ptrb, float* cf, _Float16* cfh, float* cfsq,
                             float* phalt) {
  int wave = threadIdx.x >> 5, lane = threadIdx.x & 31;
  int b = blockIdx.x * 8 + wave;
  const float* zf = af + (size_t)b * 256;
  float z[8];
#pragma unroll
  for (int i = 0; i < 8; ++i) z[i] = zf[lane * 8 + i];
  float dh = 0.f, d0 = 0.f, d1 = 0.f, d2 = 0.f;
#pragma unroll
  for (int i = 0; i < 8; ++i) {
    int c = lane * 8 + i;
    dh += z[i] * Wh[c];
    d0 += z[i] * Wsm[c];
    d1 += z[i] * Wsm[256 + c];
    d2 += z[i] * Wsm[512 + c];
  }
  dh = wsum32(dh); d0 = wsum32(d0); d1 = wsum32(d1); d2 = wsum32(d2);
  float ph = sigm(dh + bh[0]);
  d0 += bs[0]; d1 += bs[1]; d2 += bs[2];
  float mx = fmaxf(d0, fmaxf(d1, d2));
  float e0 = expf(d0 - mx), e1 = expf(d1 - mx), e2 = expf(d2 - mx);
  float es = e0 + e1 + e2;
  float push = e0 / es, pop = e1 / es, noop = e2 / es;
  float pcur = 0.f, pu = 0.f, pd = 0.f;
  if (lane < 16) {
    pcur = ptrb[b * 16 + lane];
    pu = ptrb[b * 16 + ((lane + 15) & 15)];  // roll(+1)
    pd = ptrb[b * 16 + ((lane + 1) & 15)];   // roll(-1)
  }
  float np = push * pu + pop * pd + noop * pcur;
  float s16 = wsum32(np);              // lanes 16..31 contribute 0
  np = np / (s16 + EPSF);
  float wm = push * pu;
  if (lane < 16) ptrb[b * 16 + lane] = np;
  float rd[8];
#pragma unroll
  for (int i = 0; i < 8; ++i) rd[i] = 0.f;
  float* st = stck + (size_t)b * SLOTS * 256;
  for (int s = 0; s < 16; ++s) {
    float wms = __shfl(wm, s, 32);
    float nps = __shfl(np, s, 32);
#pragma unroll
    for (int i = 0; i < 8; ++i) {
      int c = lane * 8 + i;
      float v = st[s * 256 + c];
      v = wms * z[i] + v * (1.f - wms);
      st[s * 256 + c] = v;
      rd[i] += v * nps;
    }
  }
  float sq = 0.f;
#pragma unroll
  for (int i = 0; i < 8; ++i) {
    float c = z[i] + rd[i];      // cf = concat(ar,ai) + read
    cf[(size_t)b * 256 + lane * 8 + i] = c;
    cfh[(size_t)b * 256 + lane * 8 + i] = (_Float16)c;
    sq += c * c;
  }
  sq = wsum32(sq);
  if (lane == 0) { cfsq[b] = sq; phalt[b] = ph; }
}

// ------------------------- VQ dot: cf @ codebook^T (WMMA) -------------------
__global__ void vqdot_kernel(const _Float16* cfh, const _Float16* cbT, float* dotv) {
  int wave = threadIdx.x >> 5, lane = threadIdx.x & 31;
  int m0 = blockIdx.x * 16, n0 = wave * 16;
  v8f c = vzero8();
  for (int kt = 0; kt < 8; ++kt) {
    int k = kt * 32;
    v16h a = load_a_f16(cfh, 256, m0, k);
    v16h b = load_b_f16(cbT, 128, k, n0);
    c = wmma16(a, b, c);
  }
  int r0 = (lane >> 4) << 3, nn = n0 + (lane & 15);
#pragma unroll
  for (int j = 0; j < 8; ++j)
    dotv[(size_t)(m0 + r0 + j) * NSYMS + nn] = c[j];
}

// ------------------------- per-row state update (one wave per row) ----------
__global__ void update_kernel(const float* cf, const float* cfsq, const float* dotv,
                              const float* code_sq, const float* codebook,
                              const float* adjacency, const float* phalt,
                              _Float16* zrh, _Float16* zih, float* zw,
                              float* halting, float* remain, float* vqt,
                              float* etht, float* ponder, int* cursym, int t) {
  int wave = threadIdx.x >> 5, lane = threadIdx.x & 31;
  int b = blockIdx.x * 8 + wave;
  int cur = cursym[b];
  float csq = cfsq[b];
  float bestv = 3.4e38f;
  int besti = NSYMS;
#pragma unroll
  for (int k = 0; k < 4; ++k) {
    int j = lane * 4 + k;
    float d = csq + code_sq[j] - 2.f * dotv[(size_t)b * NSYMS + j];
    if (cur >= 0) d -= GBSF * sigm(adjacency[cur * NSYMS + j]);
    if (d < bestv) { bestv = d; besti = j; }
  }
#pragma unroll
  for (int o = 16; o > 0; o >>= 1) {
    float ov = __shfl_xor(bestv, o, 32);
    int oi = __shfl_xor(besti, o, 32);
    if (ov < bestv || (ov == bestv && oi < besti)) { bestv = ov; besti = oi; }
  }
  int sym = besti;
  const float* code = codebook + (size_t)sym * 256;
  const float* c = cf + (size_t)b * 256;
  float vqa = 0.f, zn[8];
#pragma unroll
  for (int i = 0; i < 8; ++i) {
    int cc = lane * 8 + i;
    float q = code[cc], cv = c[cc];
    float dd = q - cv;
    vqa += dd * dd;
    zn[i] = 0.7f * cv + 0.3f * q;  // zq_st == zq in forward value
  }
  vqa = wsum32(vqa);
  float vq_loss = (1.f + CCF) * vqa / 256.f;
  float eth = 0.f;
  if (cur >= 0) {
    const float* arow = adjacency + (size_t)cur * NSYMS;
    float am = -3.4e38f;
#pragma unroll
    for (int k = 0; k < 4; ++k) am = fmaxf(am, arow[lane * 4 + k]);
    am = wmax32(am);
    float se = 0.f;
#pragma unroll
    for (int k = 0; k < 4; ++k) se += expf(arow[lane * 4 + k] - am);
    se = wsum32(se);
    eth = -(arow[sym] - (am + logf(se)));
  }
  float h = halting[b], r = remain[b];
  float still = (h < ACTTH) ? 1.f : 0.f;
  float p = phalt[b] * still;
  if (t == TSTEP - 1) p = r;
#pragma unroll
  for (int i = 0; i < 8; ++i) {
    int cc = lane * 8 + i;
    if (cc < 128) zrh[(size_t)b * DDIM + cc] = (_Float16)zn[i];
    else          zih[(size_t)b * DDIM + cc - 128] = (_Float16)zn[i];
    zw[(size_t)b * 256 + cc] += p * zn[i];
  }
  if (lane == 0) {
    halting[b] = h + p;
    remain[b] = r - p;
    vqt[b] += vq_loss;
    etht[b] += eth;
    ponder[b] += still;
    cursym[b] = sym;
  }
}

// ------------------------- decoder GEMM + tail ------------------------------
__global__ void logits_kernel(const float* zw, const _Float16* wdT,
                              const float* bd, float* out) {
  int wave = threadIdx.x >> 5, lane = threadIdx.x & 31;
  int m0 = blockIdx.x * 16, n0 = wave * 16;
  v8f c = vzero8();
  for (int kt = 0; kt < 8; ++kt) {
    int k = kt * 32;
    v16h a = load_a_f32(zw, 256, m0, k);
    v16h b = load_b_f16(wdT, 128, k, n0);
    c = wmma16(a, b, c);
  }
  int r0 = (lane >> 4) << 3, nn = n0 + (lane & 15);
  float bias = bd[nn];
#pragma unroll
  for (int j = 0; j < 8; ++j)
    out[(size_t)(m0 + r0 + j) * 128 + nn] = c[j] + bias;
}

__global__ void tail_kernel(const float* vqt, const float* etht,
                            const float* ponder, float* out) {
  int i = blockIdx.x * blockDim.x + threadIdx.x;
  if (i < BSZ) {
    out[i] = vqt[i];
    out[BSZ + i] = etht[i];
    out[2 * BSZ + i] = ponder[i];
  }
}

// ------------------------- host launcher ------------------------------------
extern "C" void kernel_launch(void* const* d_in, const int* in_sizes, int n_in,
                              void* d_out, int out_size, void* d_ws, size_t ws_size,
                              hipStream_t stream) {
  const int*   ids       = (const int*)d_in[0];
  const float* emb_mag   = (const float*)d_in[1];
  const float* emb_phase = (const float*)d_in[2];
  const float* cell_Wr   = (const float*)d_in[3];
  const float* cell_Wi   = (const float*)d_in[4];
  const float* ln_scale  = (const float*)d_in[5];
  const float* ln_shift  = (const float*)d_in[6];
  const float* mod_bias  = (const float*)d_in[7];
  const float* Wh        = (const float*)d_in[8];
  const float* bh        = (const float*)d_in[9];
  const float* Wsm       = (const float*)d_in[10];
  const float* bs        = (const float*)d_in[11];
  const float* Wq_r      = (const float*)d_in[12];
  const float* Wq_i      = (const float*)d_in[13];
  const float* Wk_r      = (const float*)d_in[14];
  const float* Wk_i      = (const float*)d_in[15];
  const float* Wv_r      = (const float*)d_in[16];
  const float* Wv_i      = (const float*)d_in[17];
  const float* codebook  = (const float*)d_in[18];
  const float* adjacency = (const float*)d_in[19];
  const float* Wd        = (const float*)d_in[20];
  const float* bd        = (const float*)d_in[21];

  char* w = (char*)d_ws;
  auto take = [&](size_t bytes) -> char* {
    char* p = w;
    w += (bytes + 255) & ~(size_t)255;
    return p;
  };
  _Float16* wbuf    = (_Float16*)take(196608 * sizeof(_Float16));
  float*    code_sq = (float*)take(NSYMS * 4);
  _Float16* zrh     = (_Float16*)take((size_t)BSZ * DDIM * 2);
  _Float16* zih     = (_Float16*)take((size_t)BSZ * DDIM * 2);
  _Float16* prh     = (_Float16*)take((size_t)BSZ * DDIM * 2);
  _Float16* pih     = (_Float16*)take((size_t)BSZ * DDIM * 2);
  _Float16* qh      = (_Float16*)take((size_t)BSZ * 256 * 2);
  _Float16* khT     = (_Float16*)take((size_t)BSZ * 256 * 2);
  _Float16* vh      = (_Float16*)take((size_t)BSZ * 256 * 2);
  _Float16* cfh     = (_Float16*)take((size_t)BSZ * 256 * 2);
  float*    af      = (float*)take((size_t)BSZ * 256 * 4);
  float*    cf      = (float*)take((size_t)BSZ * 256 * 4);
  float*    zw      = (float*)take((size_t)BSZ * 256 * 4);
  float*    dotv    = (float*)take((size_t)BSZ * NSYMS * 4);
  float*    stck    = (float*)take((size_t)BSZ * SLOTS * 256 * 4);
  float*    ptrb    = (float*)take((size_t)BSZ * SLOTS * 4);
  float*    halting = (float*)take(BSZ * 4);
  float*    remain  = (float*)take(BSZ * 4);
  float*    vqt     = (float*)take(BSZ * 4);
  float*    etht    = (float*)take(BSZ * 4);
  float*    ponder  = (float*)take(BSZ * 4);
  float*    phalt   = (float*)take(BSZ * 4);
  float*    cfsq    = (float*)take(BSZ * 4);
  int*      cursym  = (int*)take(BSZ * 4);

  prep_kernel<<<256, 256, 0, stream>>>(cell_Wr, cell_Wi, Wq_r, Wq_i, Wk_r, Wk_i,
                                       Wv_r, Wv_i, codebook, Wd, wbuf, code_sq);
  init_kernel<<<2048, 256, 0, stream>>>(ids, emb_mag, emb_phase, zrh, zih, zw, stck,
                                        ptrb, halting, remain, vqt, etht, ponder,
                                        cursym);
  for (int t = 0; t < TSTEP; ++t) {
    cell_kernel<<<BSZ / 16, 256, 0, stream>>>(zrh, zih, wbuf, wbuf + 16384, ln_scale,
                                              ln_shift, mod_bias, prh, pih);
    qkv_kernel<<<dim3(BSZ / 16, 3), 256, 0, stream>>>(prh, pih, wbuf, qh, khT, vh);
    attn_kernel<<<BSZ / 16, 256, 0, stream>>>(qh, khT, vh, af);
    stack_kernel<<<BSZ / 8, 256, 0, stream>>>(af, Wh, bh, Wsm, bs, stck, ptrb, cf,
                                              cfh, cfsq, phalt);
    vqdot_kernel<<<BSZ / 16, 256, 0, stream>>>(cfh, wbuf + 131072, dotv);
    update_kernel<<<BSZ / 8, 256, 0, stream>>>(cf, cfsq, dotv, code_sq, codebook,
                                               adjacency, phalt, zrh, zih, zw,
                                               halting, remain, vqt, etht, ponder,
                                               cursym, t);
  }
  logits_kernel<<<BSZ / 16, 256, 0, stream>>>(zw, wbuf + 163840, bd, (float*)d_out);
  tail_kernel<<<BSZ / 256, 256, 0, stream>>>(vqt, etht, ponder,
                                             (float*)d_out + (size_t)BSZ * 128);
}
